// GraphMAE_21345987461283
// MI455X (gfx1250) — matrix operand order repaired and checked by reading, not compiled, not run on hardware
//
#include <hip/hip_runtime.h>
#include <math.h>

// ---------------- problem constants ----------------
#define B_    8
#define N_    512
#define IN_   128
#define E_    512
#define H_    8
#define DK_   64
#define FFN_  2048
#define SV_   359     // visible tokens (512 - 153)
#define SM_   153     // masked tokens
#define SP_   512     // padded sequence length (buffers)

typedef __bf16 bf16;
typedef bf16  bf16x8 __attribute__((ext_vector_type(8)));
typedef bf16  v16bf  __attribute__((ext_vector_type(16)));
typedef float v8f    __attribute__((ext_vector_type(8)));

__device__ __forceinline__ bf16  f2b(float x) { return (bf16)x; }
__device__ __forceinline__ float b2f(bf16 x)  { return (float)x; }

// ------------- WMMA fragment loaders (CDNA5 16-bit layouts) -------------
// A 16x32 (MxK): lanes 0-15 row M=lane, VGPR0-3 K=0..7, VGPR4-7 K=16..23
//                lanes 16-31 row M=lane-16, K=8..15 and K=24..31
__device__ __forceinline__ v16bf load_a_frag(const bf16* A, int lda, int m0, int k0, int lane) {
    const bf16* p = A + (size_t)(m0 + (lane & 15)) * lda + k0 + ((lane & 16) ? 8 : 0);
    bf16x8 lo = *(const bf16x8*)p;
    bf16x8 hi = *(const bf16x8*)(p + 16);
    v16bf f;
#pragma unroll
    for (int i = 0; i < 8; ++i) { f[i] = lo[i]; f[i + 8] = hi[i]; }
    return f;
}
// B 32x16 (KxN) from Bt[n][k]: lanes 0-15 col N=lane holds K=0..7, K=16..23,
//                              lanes 16-31 col N=lane-16 holds K=8..15, K=24..31
__device__ __forceinline__ v16bf load_b_frag(const bf16* Bt, int ldb, int n0, int k0, int lane) {
    const bf16* p = Bt + (size_t)(n0 + (lane & 15)) * ldb + k0 + ((lane & 16) ? 8 : 0);
    bf16x8 lo = *(const bf16x8*)p;
    bf16x8 hi = *(const bf16x8*)(p + 16);
    v16bf f;
#pragma unroll
    for (int i = 0; i < 8; ++i) { f[i] = lo[i]; f[i + 8] = hi[i]; }
    return f;
}

__device__ __forceinline__ v8f wmma_bf16(v16bf a, v16bf b, v8f c) {
    return __builtin_amdgcn_wmma_f32_16x16x32_bf16(false, a, false, b, (short)0, c, false, false);
}

// K-loop: one wave computes a 16x64 strip (4 accumulators).  Software
// pipelined: next K-step fragments are loaded into fresh registers before the
// current WMMA chain issues, so loads overlap the matrix pipe.
__device__ __forceinline__ void wmma_kloop4(const bf16* A, int lda, const bf16* Bt, int ldb,
                                            int m0, int n0, int K, int lane, v8f acc[4]) {
    v16bf a  = load_a_frag(A, lda, m0, 0, lane);
    v16bf b0 = load_b_frag(Bt, ldb, n0,      0, lane);
    v16bf b1 = load_b_frag(Bt, ldb, n0 + 16, 0, lane);
    v16bf b2 = load_b_frag(Bt, ldb, n0 + 32, 0, lane);
    v16bf b3 = load_b_frag(Bt, ldb, n0 + 48, 0, lane);
    for (int k = 0; k < K; k += 32) {
        v16bf an = a, c0 = b0, c1 = b1, c2 = b2, c3 = b3;
        if (k + 32 < K) {
            an = load_a_frag(A, lda, m0, k + 32, lane);
            c0 = load_b_frag(Bt, ldb, n0,      k + 32, lane);
            c1 = load_b_frag(Bt, ldb, n0 + 16, k + 32, lane);
            c2 = load_b_frag(Bt, ldb, n0 + 32, k + 32, lane);
            c3 = load_b_frag(Bt, ldb, n0 + 48, k + 32, lane);
        }
        acc[0] = wmma_bf16(a, b0, acc[0]);
        acc[1] = wmma_bf16(a, b1, acc[1]);
        acc[2] = wmma_bf16(a, b2, acc[2]);
        acc[3] = wmma_bf16(a, b3, acc[3]);
        a = an; b0 = c0; b1 = c1; b2 = c2; b3 = c3;
    }
}
// C/D layout: lane<16 -> (M=r, N=lane); lane>=16 -> (M=r+8, N=lane-16)

// ---------------- generic linear-layer GEMM ----------------
// C[z][m][n] = sum_k A[z][m][k] * Bt[n][k]  (+bias[n]) (+gelu) (+Res)
// wave -> 16x64 strip; block (8 waves) -> 16x512 strip
__global__ __launch_bounds__(256)
void gemm_bf16_kernel(const bf16* __restrict__ A, long long sA, int lda,
                      const bf16* __restrict__ Bt, long long sB, int ldb,
                      const float* __restrict__ bias,
                      float* __restrict__ Cf, bf16* __restrict__ Cb,
                      long long sC, int ldc,
                      const float* __restrict__ Res,
                      int M, int N, int K, int act) {
    int lane = threadIdx.x & 31;
    int wave = threadIdx.x >> 5;
    int n0 = (blockIdx.x * 8 + wave) * 64;
    int m0 = blockIdx.y * 16;
    if (n0 >= N) return;
    long long z = blockIdx.z;
    v8f acc[4];
#pragma unroll
    for (int t = 0; t < 4; ++t)
#pragma unroll
        for (int i = 0; i < 8; ++i) acc[t][i] = 0.0f;
    wmma_kloop4(A + z * sA, lda, Bt + z * sB, ldb, m0, n0, K, lane, acc);
    int mb = m0 + ((lane & 16) ? 8 : 0);
#pragma unroll
    for (int t = 0; t < 4; ++t) {
        int nn = n0 + 16 * t + (lane & 15);
        if (nn >= N) continue;
        float bv = bias ? bias[nn] : 0.0f;
#pragma unroll
        for (int r = 0; r < 8; ++r) {
            int mm = mb + r;
            if (mm < M) {
                float v = acc[t][r] + bv;
                if (act) v = 0.5f * v * (1.0f + erff(v * 0.70710678118654752f));  // exact GELU
                long long ci = z * sC + (long long)mm * ldc + nn;
                if (Res) v += Res[ci];
                if (Cf) Cf[ci] = v;
                if (Cb) Cb[ci] = f2b(v);
            }
        }
    }
}

// ---------------- attention scores:  s = (Q K^T)/sqrt(dk) + gathered bias ----------------
// wave -> 16 queries x 64 keys
__global__ __launch_bounds__(32)
void scores_kernel(const bf16* __restrict__ Q, const bf16* __restrict__ Kb,
                   const float* __restrict__ attn_bias,
                   const int* __restrict__ spatial_pos,
                   const float* __restrict__ spatial_emb,
                   const int* __restrict__ idx,
                   float* __restrict__ Sc, int S, float scale) {
    int lane = threadIdx.x & 31;
    int n0 = blockIdx.x * 64, m0 = blockIdx.y * 16;
    int z = blockIdx.z, b = z / H_, h = z % H_;
    const bf16* Qz = Q  + (size_t)b * SP_ * E_ + h * DK_;
    const bf16* Kz = Kb + (size_t)b * SP_ * E_ + h * DK_;
    v8f acc[4];
#pragma unroll
    for (int t = 0; t < 4; ++t)
#pragma unroll
        for (int i = 0; i < 8; ++i) acc[t][i] = 0.0f;
    wmma_kloop4(Qz, E_, Kz, E_, m0, n0, DK_, lane, acc);
    int mb = m0 + ((lane & 16) ? 8 : 0);
    float* out = Sc + (size_t)z * SP_ * SP_;
#pragma unroll
    for (int t = 0; t < 4; ++t) {
        int j = n0 + 16 * t + (lane & 15);
        if (j >= S) continue;
        int gj = idx[j];
#pragma unroll
        for (int r = 0; r < 8; ++r) {
            int i = mb + r;
            if (i < S) {
                int gi = idx[i];
                size_t ij = ((size_t)b * N_ + gi) * N_ + gj;
                float bias = 2.0f * attn_bias[ij] + spatial_emb[spatial_pos[ij] * H_ + h];
                out[(size_t)i * SP_ + j] = acc[t][r] * scale + bias;
            }
        }
    }
}

// ---------------- attention output:  O = P V  (P bf16, Vt pre-transposed per head) ----------------
// wave -> 16 queries x full dk=64
__global__ __launch_bounds__(32)
void av_kernel(const bf16* __restrict__ P, const bf16* __restrict__ Vt,
               bf16* __restrict__ O, int S) {
    int lane = threadIdx.x & 31;
    int m0 = blockIdx.x * 16;   // query tile
    int z = blockIdx.y, b = z / H_, h = z % H_;
    v8f acc[4];
#pragma unroll
    for (int t = 0; t < 4; ++t)
#pragma unroll
        for (int i = 0; i < 8; ++i) acc[t][i] = 0.0f;
    wmma_kloop4(P + (size_t)z * SP_ * SP_, SP_,
                Vt + (size_t)z * DK_ * SP_, SP_, m0, 0, SP_, lane, acc);
    int mb = m0 + ((lane & 16) ? 8 : 0);
    bf16* Ob = O + (size_t)b * SP_ * E_ + h * DK_;
#pragma unroll
    for (int t = 0; t < 4; ++t) {
        int d = 16 * t + (lane & 15);
#pragma unroll
        for (int r = 0; r < 8; ++r) {
            int s = mb + r;
            if (s < S) Ob[(size_t)s * E_ + d] = f2b(acc[t][r]);
        }
    }
}

// ---------------- layernorm (f32 in -> bf16 out) ----------------
__global__ __launch_bounds__(128)
void ln_kernel(const float* __restrict__ X, const float* __restrict__ w,
               const float* __restrict__ bb, bf16* __restrict__ Y) {
    __shared__ float red[128];
    int s = blockIdx.x, b = blockIdx.y, t = threadIdx.x;
    const float* xr = X + ((size_t)b * SP_ + s) * E_;
    float vals[4], lsum = 0.f, lsq = 0.f;
#pragma unroll
    for (int i = 0; i < 4; ++i) { float v = xr[t + i * 128]; vals[i] = v; lsum += v; lsq += v * v; }
    red[t] = lsum; __syncthreads();
    for (int o = 64; o > 0; o >>= 1) { if (t < o) red[t] += red[t + o]; __syncthreads(); }
    float mean = red[0] / E_; __syncthreads();
    red[t] = lsq; __syncthreads();
    for (int o = 64; o > 0; o >>= 1) { if (t < o) red[t] += red[t + o]; __syncthreads(); }
    float var = red[0] / E_ - mean * mean;
    float rstd = rsqrtf(var + 1e-5f);
    bf16* yr = Y + ((size_t)b * SP_ + s) * E_;
#pragma unroll
    for (int i = 0; i < 4; ++i) {
        int e = t + i * 128;
        yr[e] = f2b((vals[i] - mean) * rstd * w[e] + bb[e]);
    }
}

// ---------------- softmax over valid keys, zero-fill padded keys ----------------
__global__ __launch_bounds__(256)
void softmax_kernel(const float* __restrict__ Sc, bf16* __restrict__ P, int S) {
    __shared__ float red[256];
    int i = blockIdx.x, z = blockIdx.y, t = threadIdx.x;
    const float* row = Sc + ((size_t)z * SP_ + i) * SP_;
    bf16* pr = P + ((size_t)z * SP_ + i) * SP_;
    float mx = -3.4e38f;
    for (int j = t; j < S; j += 256) mx = fmaxf(mx, row[j]);
    red[t] = mx; __syncthreads();
    for (int o = 128; o > 0; o >>= 1) { if (t < o) red[t] = fmaxf(red[t], red[t + o]); __syncthreads(); }
    mx = red[0]; __syncthreads();
    float sum = 0.f;
    for (int j = t; j < S; j += 256) sum += __expf(row[j] - mx);
    red[t] = sum; __syncthreads();
    for (int o = 128; o > 0; o >>= 1) { if (t < o) red[t] += red[t + o]; __syncthreads(); }
    float inv = 1.0f / red[0];
    for (int j = t; j < SP_; j += 256)
        pr[j] = (j < S) ? f2b(__expf(row[j] - mx) * inv) : f2b(0.0f);
}

// ---------------- small elementwise / prep kernels ----------------
__global__ void f2b_kernel(const float* __restrict__ X, bf16* __restrict__ Y, long long n) {
    long long i = (long long)blockIdx.x * 256 + threadIdx.x;
    if (i < n) Y[i] = f2b(X[i]);
}
// Wt[n][k] = bf16(W[k][n])  (weights stored K-major in reference)
__global__ void conv_t_kernel(const float* __restrict__ W, bf16* __restrict__ Wt, int K, int N) {
    long long i = (long long)blockIdx.x * 256 + threadIdx.x;
    if (i >= (long long)K * N) return;
    int n = (int)(i / K), k = (int)(i % K);
    Wt[i] = f2b(W[(long long)k * N + n]);
}
// feat += in_deg_emb[in_degree] + out_deg_emb[in_degree]   (reference uses in_degree twice)
__global__ void embed_add_kernel(float* __restrict__ feat, const int* __restrict__ indeg,
                                 const float* __restrict__ ide, const float* __restrict__ ode) {
    long long i = (long long)blockIdx.x * 256 + threadIdx.x;
    if (i >= (long long)B_ * N_ * E_) return;
    int e = (int)(i % E_);
    long long bn = i / E_;
    int dg = indeg[bn];
    feat[i] += ide[(size_t)dg * E_ + e] + ode[(size_t)dg * E_ + e];
}
__global__ void gather_rows_kernel(const float* __restrict__ feat, const int* __restrict__ vis,
                                   float* __restrict__ X) {
    long long i = (long long)blockIdx.x * 256 + threadIdx.x;
    if (i >= (long long)B_ * SV_ * E_) return;
    int e = (int)(i % E_);
    long long t = i / E_;
    int s = (int)(t % SV_), b = (int)(t / SV_);
    X[((size_t)b * SP_ + s) * E_ + e] = feat[((size_t)b * N_ + vis[s]) * E_ + e];
}
__global__ void concat_idx_kernel(const int* __restrict__ vis, const int* __restrict__ msk,
                                  int* __restrict__ out) {
    int i = blockIdx.x * 256 + threadIdx.x;
    if (i < SV_) out[i] = vis[i];
    else if (i < SP_) out[i] = msk[i - SV_];
}
// Vt[b][h][d][t] = V[b][t][h*64+d]; zero for t>=S so padded probs contribute 0
__global__ void transpose_v_kernel(const bf16* __restrict__ V, bf16* __restrict__ Vt, int S) {
    long long i = (long long)blockIdx.x * 256 + threadIdx.x;
    if (i >= (long long)B_ * H_ * DK_ * SP_) return;
    int t = (int)(i % SP_);
    long long r = i / SP_;
    int d = (int)(r % DK_);
    long long r2 = r / DK_;
    int h = (int)(r2 % H_), b = (int)(r2 / H_);
    Vt[i] = (t < S) ? V[((size_t)b * SP_ + t) * E_ + h * DK_ + d] : f2b(0.0f);
}
// d = concat(h+pos[vis], mask_token+pos[mask]);  pos = ide[in_deg] + ode[out_deg]
__global__ void build_dec_kernel(const float* __restrict__ hdec, const float* __restrict__ mtok,
                                 const int* __restrict__ vis, const int* __restrict__ msk,
                                 const int* __restrict__ indeg, const int* __restrict__ outdeg,
                                 const float* __restrict__ ide, const float* __restrict__ ode,
                                 float* __restrict__ D) {
    long long i = (long long)blockIdx.x * 256 + threadIdx.x;
    if (i >= (long long)B_ * SP_ * E_) return;
    int e = (int)(i % E_);
    long long t = i / E_;
    int s = (int)(t % SP_), b = (int)(t / SP_);
    int n = (s < SV_) ? vis[s] : msk[s - SV_];
    float pos = ide[(size_t)indeg[(size_t)b * N_ + n] * E_ + e] +
                ode[(size_t)outdeg[(size_t)b * N_ + n] * E_ + e];
    float base = (s < SV_) ? hdec[i] : mtok[e];
    D[i] = base + pos;
}

// =====================================================================
extern "C" void kernel_launch(void* const* d_in, const int* in_sizes, int n_in,
                              void* d_out, int out_size, void* d_ws, size_t ws_size,
                              hipStream_t stream) {
    (void)in_sizes; (void)n_in; (void)out_size; (void)ws_size;

    // ---- inputs (setup_inputs dict order, nested dicts flattened in insertion order) ----
    const float* x      = (const float*)d_in[0];
    const float* attnb  = (const float*)d_in[1];
    const int*   spos   = (const int*)d_in[2];
    const int*   indeg  = (const int*)d_in[3];
    const int*   outdeg = (const int*)d_in[4];
    const int*   vis    = (const int*)d_in[5];
    const int*   msk    = (const int*)d_in[6];
    const float* ipw    = (const float*)d_in[7];
    const float* ipb    = (const float*)d_in[8];
    const float* semb   = (const float*)d_in[9];
    const float* ide    = (const float*)d_in[10];
    const float* ode    = (const float*)d_in[11];
    // enc_params: 12..27 -> ln1_w ln1_b wq bq wk bk wv bv wo bo ln2_w ln2_b w1 b1 w2 b2
    const float* enc_ln_w = (const float*)d_in[28];
    const float* enc_ln_b = (const float*)d_in[29];
    const float* e2dw     = (const float*)d_in[30];
    const float* mtok     = (const float*)d_in[31];
    // dec_params: 32..47
    const float* dec_ln_w = (const float*)d_in[48];
    const float* dec_ln_b = (const float*)d_in[49];
    const float* outw     = (const float*)d_in[50];
    const float* outb     = (const float*)d_in[51];

    // ---- workspace carve-up ----
    char* base = (char*)d_ws;
    size_t off = 0;
    auto alloc = [&](size_t bytes) -> void* {
        void* p = base + off;
        off += (bytes + 255) & ~(size_t)255;
        return p;
    };
    struct WT { bf16 *wq, *wk, *wv, *wo, *w1, *w2; };
    bf16* ipwt = (bf16*)alloc((size_t)E_ * IN_ * 2);
    WT encW[4], decW[2];
    for (int i = 0; i < 4; ++i) {
        encW[i].wq = (bf16*)alloc((size_t)E_ * E_ * 2);
        encW[i].wk = (bf16*)alloc((size_t)E_ * E_ * 2);
        encW[i].wv = (bf16*)alloc((size_t)E_ * E_ * 2);
        encW[i].wo = (bf16*)alloc((size_t)E_ * E_ * 2);
        encW[i].w1 = (bf16*)alloc((size_t)FFN_ * E_ * 2);
        encW[i].w2 = (bf16*)alloc((size_t)E_ * FFN_ * 2);
    }
    bf16* e2dt = (bf16*)alloc((size_t)E_ * E_ * 2);
    for (int i = 0; i < 2; ++i) {
        decW[i].wq = (bf16*)alloc((size_t)E_ * E_ * 2);
        decW[i].wk = (bf16*)alloc((size_t)E_ * E_ * 2);
        decW[i].wv = (bf16*)alloc((size_t)E_ * E_ * 2);
        decW[i].wo = (bf16*)alloc((size_t)E_ * E_ * 2);
        decW[i].w1 = (bf16*)alloc((size_t)FFN_ * E_ * 2);
        decW[i].w2 = (bf16*)alloc((size_t)E_ * FFN_ * 2);
    }
    bf16*  outwt  = (bf16*)alloc((size_t)IN_ * E_ * 2);
    float* feat   = (float*)alloc((size_t)B_ * N_ * E_ * 4);
    float* xres   = (float*)alloc((size_t)B_ * SP_ * E_ * 4);
    float* hdec   = (float*)alloc((size_t)B_ * SP_ * E_ * 4);
    float* scores = (float*)alloc((size_t)B_ * H_ * SP_ * SP_ * 4);
    bf16*  xb     = (bf16*)alloc((size_t)B_ * N_ * IN_ * 2);
    bf16*  ybf    = (bf16*)alloc((size_t)B_ * SP_ * E_ * 2);
    bf16*  qb     = (bf16*)alloc((size_t)B_ * SP_ * E_ * 2);
    bf16*  kb     = (bf16*)alloc((size_t)B_ * SP_ * E_ * 2);
    bf16*  vb     = (bf16*)alloc((size_t)B_ * SP_ * E_ * 2);
    bf16*  vt     = (bf16*)alloc((size_t)B_ * H_ * DK_ * SP_ * 2);
    bf16*  pb     = (bf16*)alloc((size_t)B_ * H_ * SP_ * SP_ * 2);
    bf16*  attnbf = (bf16*)alloc((size_t)B_ * SP_ * E_ * 2);
    bf16*  midb   = (bf16*)alloc((size_t)B_ * SP_ * FFN_ * 2);
    int*   nidx   = (int*)alloc((size_t)SP_ * 4);
    (void)alloc(1 << 16);  // tail pad: over-read slack for ragged M tiles

    // ---- helpers ----
    auto convt = [&](const float* W, bf16* Wt, int K, int N) {
        long long tot = (long long)K * N;
        conv_t_kernel<<<dim3((unsigned)((tot + 255) / 256)), 256, 0, stream>>>(W, Wt, K, N);
    };
    auto gemm = [&](const bf16* A, long long sA, int lda, const bf16* Bt, long long sB, int ldb,
                    const float* bias, float* Cf, bf16* Cb, long long sC, int ldc,
                    const float* Res, int M, int N, int K, int Z, int act) {
        dim3 g((unsigned)((N + 511) / 512), (unsigned)((M + 15) / 16), (unsigned)Z);
        gemm_bf16_kernel<<<g, 256, 0, stream>>>(A, sA, lda, Bt, sB, ldb, bias, Cf, Cb, sC, ldc,
                                                Res, M, N, K, act);
    };

    // ---- one-time weight conversion (deterministic, re-done every launch) ----
    convt(ipw, ipwt, IN_, E_);
    for (int i = 0; i < 4; ++i) {
        convt((const float*)d_in[14] + (size_t)i * E_ * E_,  encW[i].wq, E_, E_);
        convt((const float*)d_in[16] + (size_t)i * E_ * E_,  encW[i].wk, E_, E_);
        convt((const float*)d_in[18] + (size_t)i * E_ * E_,  encW[i].wv, E_, E_);
        convt((const float*)d_in[20] + (size_t)i * E_ * E_,  encW[i].wo, E_, E_);
        convt((const float*)d_in[24] + (size_t)i * E_ * FFN_, encW[i].w1, E_, FFN_);
        convt((const float*)d_in[26] + (size_t)i * FFN_ * E_, encW[i].w2, FFN_, E_);
    }
    convt(e2dw, e2dt, E_, E_);
    for (int i = 0; i < 2; ++i) {
        convt((const float*)d_in[34] + (size_t)i * E_ * E_,  decW[i].wq, E_, E_);
        convt((const float*)d_in[36] + (size_t)i * E_ * E_,  decW[i].wk, E_, E_);
        convt((const float*)d_in[38] + (size_t)i * E_ * E_,  decW[i].wv, E_, E_);
        convt((const float*)d_in[40] + (size_t)i * E_ * E_,  decW[i].wo, E_, E_);
        convt((const float*)d_in[44] + (size_t)i * E_ * FFN_, decW[i].w1, E_, FFN_);
        convt((const float*)d_in[46] + (size_t)i * FFN_ * E_, decW[i].w2, FFN_, E_);
    }
    convt(outw, outwt, E_, IN_);

    const long long SPE = (long long)SP_ * E_;
    const long long SPF = (long long)SP_ * FFN_;

    // ---- embedding: feat = x @ input_proj_w + b + deg embeddings; gather vis rows ----
    {
        long long nx = (long long)B_ * N_ * IN_;
        f2b_kernel<<<dim3((unsigned)((nx + 255) / 256)), 256, 0, stream>>>(x, xb, nx);
        gemm(xb, 0, IN_, ipwt, 0, IN_, ipb, feat, nullptr, 0, E_, nullptr, B_ * N_, E_, IN_, 1, 0);
        long long nf = (long long)B_ * N_ * E_;
        embed_add_kernel<<<dim3((unsigned)((nf + 255) / 256)), 256, 0, stream>>>(feat, indeg, ide, ode);
        long long ng = (long long)B_ * SV_ * E_;
        gather_rows_kernel<<<dim3((unsigned)((ng + 255) / 256)), 256, 0, stream>>>(feat, vis, xres);
        concat_idx_kernel<<<2, 256, 0, stream>>>(vis, msk, nidx);
    }

    // ---- one transformer layer ----
    struct LP { const float *ln1w, *ln1b, *bq, *bk, *bv, *bo, *ln2w, *ln2b, *b1, *b2; };
    auto run_layer = [&](const LP& p, const WT& w, int S, const int* idxa) {
        int st = (S + 15) / 16;
        ln_kernel<<<dim3(S, B_), 128, 0, stream>>>(xres, p.ln1w, p.ln1b, ybf);
        gemm(ybf, SPE, E_, w.wq, 0, E_, p.bq, nullptr, qb, SPE, E_, nullptr, S, E_, E_, B_, 0);
        gemm(ybf, SPE, E_, w.wk, 0, E_, p.bk, nullptr, kb, SPE, E_, nullptr, S, E_, E_, B_, 0);
        gemm(ybf, SPE, E_, w.wv, 0, E_, p.bv, nullptr, vb, SPE, E_, nullptr, S, E_, E_, B_, 0);
        long long nv = (long long)B_ * H_ * DK_ * SP_;
        transpose_v_kernel<<<dim3((unsigned)((nv + 255) / 256)), 256, 0, stream>>>(vb, vt, S);
        scores_kernel<<<dim3((S + 63) / 64, st, B_ * H_), 32, 0, stream>>>(qb, kb, attnb, spos,
                                                                           semb, idxa, scores, S,
                                                                           0.125f);
        softmax_kernel<<<dim3(S, B_ * H_), 256, 0, stream>>>(scores, pb, S);
        av_kernel<<<dim3(st, B_ * H_), 32, 0, stream>>>(pb, vt, attnbf, S);
        gemm(attnbf, SPE, E_, w.wo, 0, E_, p.bo, xres, nullptr, SPE, E_, xres, S, E_, E_, B_, 0);
        ln_kernel<<<dim3(S, B_), 128, 0, stream>>>(xres, p.ln2w, p.ln2b, ybf);
        gemm(ybf, SPE, E_, w.w1, 0, E_, p.b1, nullptr, midb, SPF, FFN_, nullptr, S, FFN_, E_, B_, 1);
        gemm(midb, SPF, FFN_, w.w2, 0, FFN_, p.b2, xres, nullptr, SPE, E_, xres, S, E_, FFN_, B_, 0);
    };

    // ---- encoder (4 layers, S = 359) ----
    for (int i = 0; i < 4; ++i) {
        LP p = { (const float*)d_in[12] + (size_t)i * E_, (const float*)d_in[13] + (size_t)i * E_,
                 (const float*)d_in[15] + (size_t)i * E_, (const float*)d_in[17] + (size_t)i * E_,
                 (const float*)d_in[19] + (size_t)i * E_, (const float*)d_in[21] + (size_t)i * E_,
                 (const float*)d_in[22] + (size_t)i * E_, (const float*)d_in[23] + (size_t)i * E_,
                 (const float*)d_in[25] + (size_t)i * FFN_, (const float*)d_in[27] + (size_t)i * E_ };
        run_layer(p, encW[i], SV_, vis);
    }

    // ---- encoder final LN + enc2dec projection ----
    ln_kernel<<<dim3(SV_, B_), 128, 0, stream>>>(xres, enc_ln_w, enc_ln_b, ybf);
    gemm(ybf, SPE, E_, e2dt, 0, E_, nullptr, hdec, nullptr, SPE, E_, nullptr, SV_, E_, E_, B_, 0);

    // ---- build decoder input ----
    {
        long long nd = (long long)B_ * SP_ * E_;
        build_dec_kernel<<<dim3((unsigned)((nd + 255) / 256)), 256, 0, stream>>>(
            hdec, mtok, vis, msk, indeg, outdeg, ide, ode, xres);
    }

    // ---- decoder (2 layers, S = 512) ----
    for (int i = 0; i < 2; ++i) {
        LP p = { (const float*)d_in[32] + (size_t)i * E_, (const float*)d_in[33] + (size_t)i * E_,
                 (const float*)d_in[35] + (size_t)i * E_, (const float*)d_in[37] + (size_t)i * E_,
                 (const float*)d_in[39] + (size_t)i * E_, (const float*)d_in[41] + (size_t)i * E_,
                 (const float*)d_in[42] + (size_t)i * E_, (const float*)d_in[43] + (size_t)i * E_,
                 (const float*)d_in[45] + (size_t)i * FFN_, (const float*)d_in[47] + (size_t)i * E_ };
        run_layer(p, decW[i], SP_, nidx);
    }

    // ---- final LN + output projection on last 153 (masked) rows ----
    ln_kernel<<<dim3(SP_, B_), 128, 0, stream>>>(xres, dec_ln_w, dec_ln_b, ybf);
    gemm(ybf + (size_t)SV_ * E_, SPE, E_, outwt, 0, E_, outb,
         (float*)d_out, nullptr, (long long)SM_ * IN_, IN_, nullptr, SM_, IN_, E_, B_, 0);
}
